// ChebNet_15358803050746
// MI455X (gfx1250) — matrix operand order repaired and checked by reading, not compiled, not run on hardware
//
#include <hip/hip_runtime.h>
#include <hip/hip_bf16.h>

// ---------------------------------------------------------------------------
// ChebNet (K=5, HID=64) for MI455X / gfx1250.
//  - edge prop: float4 gather * norm -> global_atomic_add_f32 scatter
//  - dense Tx @ W: v_wmma_f32_16x16x32_bf16, W^T staged in LDS as bf16
//    (B: contiguous per-lane fragments -> ds_load_b128;
//     A: unconditional row-clamped global_load_b128 + v_cvt_pk_bf16_f32)
// ---------------------------------------------------------------------------

typedef __attribute__((ext_vector_type(16))) __bf16 bf16x16;
typedef __attribute__((ext_vector_type(8)))  __bf16 bf16x8;
typedef __attribute__((ext_vector_type(8)))  float  floatx8;

#define FEAT 64
#define WT_STRIDE 72   // padded bf16 column stride: 144 B = 9*16 B (b128-aligned)

// ---------------- elementwise / setup kernels ------------------------------

__global__ void k_zero(float* __restrict__ p, long long n) {
  long long i = (long long)blockIdx.x * blockDim.x + threadIdx.x;
  long long stride = (long long)gridDim.x * blockDim.x;
  for (; i < n; i += stride) p[i] = 0.0f;
}

__global__ void k_deg(const long long* __restrict__ src, float* __restrict__ deg,
                      long long E) {
  long long e = (long long)blockIdx.x * blockDim.x + threadIdx.x;
  if (e < E) atomicAdd(&deg[src[e]], 1.0f);
}

__global__ void k_norm(const long long* __restrict__ src,
                       const long long* __restrict__ dst,
                       const float* __restrict__ deg,
                       float* __restrict__ nrm, long long E) {
  long long e = (long long)blockIdx.x * blockDim.x + threadIdx.x;
  if (e < E) {
    float ds = deg[src[e]], dd = deg[dst[e]];
    float is = ds > 0.0f ? rsqrtf(fmaxf(ds, 1e-12f)) : 0.0f;
    float id = dd > 0.0f ? rsqrtf(fmaxf(dd, 1e-12f)) : 0.0f;
    nrm[e] = -(is * id);
  }
}

__global__ void k_negcopy(float* __restrict__ out, const float* __restrict__ in,
                          long long n) {
  long long i = (long long)blockIdx.x * blockDim.x + threadIdx.x;
  if (i < n) out[i] = -in[i];
}

// out[dst] += scale * norm[e] * h[src]   (F = 1)
__global__ void k_prop1(const float* __restrict__ h,
                        const long long* __restrict__ src,
                        const long long* __restrict__ dst,
                        const float* __restrict__ nrm,
                        float* __restrict__ out, float scale, long long E) {
  long long e = (long long)blockIdx.x * blockDim.x + threadIdx.x;
  if (e < E) {
    long long s = src[e], d = dst[e];
    atomicAdd(&out[d], scale * nrm[e] * h[s]);
  }
}

// out[dst, 4q..4q+3] += scale * norm[e] * h[src, 4q..4q+3]
// one thread per (edge, quad): float4 gather, 4x atomic f32 scatter
__global__ void k_propF(const float* __restrict__ h,
                        const long long* __restrict__ src,
                        const long long* __restrict__ dst,
                        const float* __restrict__ nrm,
                        float* __restrict__ out, float scale, long long EQ) {
  long long t = (long long)blockIdx.x * blockDim.x + threadIdx.x;
  if (t < EQ) {
    long long e = t >> 4;
    int q = (int)(t & 15);
    long long s = src[e], d = dst[e];
    float w = scale * nrm[e];
    const float4 v = *(const float4*)(h + s * FEAT + q * 4);
    float* o = out + d * FEAT + q * 4;
    atomicAdd(o + 0, w * v.x);
    atomicAdd(o + 1, w * v.y);
    atomicAdd(o + 2, w * v.z);
    atomicAdd(o + 3, w * v.w);
  }
}

// acc[n,f] += tx[n] * w[f]    (layer-1 rank-1 GEMM, Fin = 1)
__global__ void k_rank1(float* __restrict__ acc, const float* __restrict__ tx,
                        const float* __restrict__ w, long long NF) {
  long long t = (long long)blockIdx.x * blockDim.x + threadIdx.x;
  if (t < NF) {
    long long n = t >> 6;
    int f = (int)(t & 63);
    acc[t] += tx[n] * w[f];
  }
}

__global__ void k_bias_relu(const float* __restrict__ acc,
                            const float* __restrict__ b,
                            float* __restrict__ h, long long NF) {
  long long t = (long long)blockIdx.x * blockDim.x + threadIdx.x;
  if (t < NF) {
    float v = acc[t] + b[t & 63];
    h[t] = v > 0.0f ? v : 0.0f;
  }
}

__global__ void k_fc(const float* __restrict__ h, const float* __restrict__ wfc,
                     const float* __restrict__ bfc, float* __restrict__ out,
                     int N) {
  int n = blockIdx.x * blockDim.x + threadIdx.x;
  if (n < N) {
    float s = 0.0f;
    const float* row = h + (long long)n * FEAT;
#pragma unroll
    for (int f = 0; f < FEAT; ++f) s += row[f] * wfc[f];
    out[n] = s + bfc[0];
  }
}

// ---------------- WMMA GEMM:  acc[N,64] += A[N,64] @ Wk[64,64] -------------
// 256 threads = 8 waves; each wave owns a 16-row x 64-col output stripe.
// Per wave: 2 k-chunks (K=32) x 4 n-tiles -> 8 v_wmma_f32_16x16x32_bf16.
// Wk is staged TRANSPOSED (bf16) in LDS so every B fragment per lane is 16
// contiguous bf16 -> two ds_load_b128, no scalar packing.
// A rows are clamped (not predicated) so loads stay vector & unconditional.

__launch_bounds__(256)
__global__ void k_gemm_wmma(const float* __restrict__ A,
                            const float* __restrict__ Wk,
                            float* __restrict__ acc, int N) {
  __shared__ __align__(16) __bf16 sWt[FEAT * WT_STRIDE];  // ~9 KB
  for (int i = threadIdx.x; i < FEAT * FEAT; i += blockDim.x) {
    const int kk = i >> 6, nn = i & 63;
    sWt[nn * WT_STRIDE + kk] = (__bf16)Wk[i];  // sWt[col][k] = W[k][col]
  }
  __syncthreads();

  const int wave = threadIdx.x >> 5;
  const int lane = threadIdx.x & 31;
  const int half = lane >> 4;   // 0: lanes 0-15, 1: lanes 16-31
  const int m    = lane & 15;
  const int rowBase = blockIdx.x * 128 + wave * 16;
  if (rowBase >= N) return;     // wave-uniform exit (no barriers after this)

  floatx8 c0 = {}, c1 = {}, c2 = {}, c3 = {};

  // Clamp instead of predicate: all A loads unconditional vector loads.
  // Out-of-range lanes duplicate the last row; their outputs are not stored.
  int row = rowBase + m;
  row = row < N ? row : (N - 1);
  const float* arow = A + (long long)row * FEAT;

#pragma unroll
  for (int ch = 0; ch < 2; ++ch) {
    const int kc = ch * 32;

    // A fragment 16x32 bf16: lane l = row (l&15);
    // elements 0..7  -> K = kc + half*16 + (0..7)   (contiguous)
    // elements 8..15 -> K = kc + half*16 + 8 + (0..7) (contiguous)
    const float* ap0 = arow + kc + half * 16;
    const float* ap1 = arow + kc + half * 16 + 8;
    const float4 f0 = *(const float4*)(ap0 + 0);
    const float4 f1 = *(const float4*)(ap0 + 4);
    const float4 f2 = *(const float4*)(ap1 + 0);
    const float4 f3 = *(const float4*)(ap1 + 4);
    bf16x16 a;
    a[0]  = (__bf16)f0.x; a[1]  = (__bf16)f0.y; a[2]  = (__bf16)f0.z; a[3]  = (__bf16)f0.w;
    a[4]  = (__bf16)f1.x; a[5]  = (__bf16)f1.y; a[6]  = (__bf16)f1.z; a[7]  = (__bf16)f1.w;
    a[8]  = (__bf16)f2.x; a[9]  = (__bf16)f2.y; a[10] = (__bf16)f2.z; a[11] = (__bf16)f2.w;
    a[12] = (__bf16)f3.x; a[13] = (__bf16)f3.y; a[14] = (__bf16)f3.z; a[15] = (__bf16)f3.w;

    // B fragments 32x16 bf16: lane l = column (l&15);
    // element i -> K = kc + half*16 + i  => 16 contiguous bf16 in sWt column
#pragma unroll
    for (int j = 0; j < 4; ++j) {
      const __bf16* bp = &sWt[(j * 16 + m) * WT_STRIDE + kc + half * 16];
      const bf16x8 blo = *(const bf16x8*)(bp);
      const bf16x8 bhi = *(const bf16x8*)(bp + 8);
      const bf16x16 b = __builtin_shufflevector(
          blo, bhi, 0, 1, 2, 3, 4, 5, 6, 7, 8, 9, 10, 11, 12, 13, 14, 15);
      if (j == 0)
        c0 = __builtin_amdgcn_wmma_f32_16x16x32_bf16(false, a, false, b,
                                                     (short)0, c0, false, false);
      else if (j == 1)
        c1 = __builtin_amdgcn_wmma_f32_16x16x32_bf16(false, a, false, b,
                                                     (short)0, c1, false, false);
      else if (j == 2)
        c2 = __builtin_amdgcn_wmma_f32_16x16x32_bf16(false, a, false, b,
                                                     (short)0, c2, false, false);
      else
        c3 = __builtin_amdgcn_wmma_f32_16x16x32_bf16(false, a, false, b,
                                                     (short)0, c3, false, false);
    }
  }

  // C/D layout: VGPR r holds M = r + 8*half, N = lane&15.
#pragma unroll
  for (int r = 0; r < 8; ++r) {
    const int orowi = rowBase + r + 8 * half;
    if (orowi < N) {
      float* orow = acc + (long long)orowi * FEAT;
      orow[ 0 + m] += c0[r];
      orow[16 + m] += c1[r];
      orow[32 + m] += c2[r];
      orow[48 + m] += c3[r];
    }
  }
}

// ---------------------------------------------------------------------------

static inline int nblk(long long n, int t) { return (int)((n + t - 1) / t); }

extern "C" void kernel_launch(void* const* d_in, const int* in_sizes, int n_in,
                              void* d_out, int out_size, void* d_ws, size_t ws_size,
                              hipStream_t stream) {
  const float*     x    = (const float*)d_in[0];
  const long long* ei   = (const long long*)d_in[1];  // int64 [2,E]
  const float*     W1   = (const float*)d_in[2];      // [5,1,64]
  const float*     b1   = (const float*)d_in[3];
  const float*     W2   = (const float*)d_in[4];      // [5,64,64]
  const float*     b2   = (const float*)d_in[5];
  const float*     W3   = (const float*)d_in[6];      // [5,64,64]
  const float*     b3   = (const float*)d_in[7];
  const float*     Wfc  = (const float*)d_in[8];      // [64,1]
  const float*     bfc  = (const float*)d_in[9];

  const int       N  = in_sizes[0];            // x is [N,1]
  const long long E  = (long long)in_sizes[1] / 2;
  const long long NF = (long long)N * FEAT;
  const long long EQ = E * 16;                 // (edge, quad) work items
  const long long* srcI = ei;
  const long long* dstI = ei + E;

  // workspace carve (floats): deg[N] nrm[E] h[NF] bufA[NF] bufB[NF] bufC[NF] acc[NF]
  float* ws   = (float*)d_ws;
  float* deg  = ws;                 size_t off = (size_t)N;
  float* nrm  = ws + off;           off += (size_t)E;
  float* h    = ws + off;           off += (size_t)NF;
  float* bufA = ws + off;           off += (size_t)NF;
  float* bufB = ws + off;           off += (size_t)NF;
  float* bufC = ws + off;           off += (size_t)NF;
  float* acc  = ws + off;           off += (size_t)NF;

  const int T = 256;
  const int gemmGrid = (N + 127) / 128;

  // ---- normalization: deg over src, norm = -dinv[src]*dinv[dst] ----
  k_zero<<<nblk(N, T), T, 0, stream>>>(deg, N);
  k_deg<<<nblk(E, T), T, 0, stream>>>(srcI, deg, E);
  k_norm<<<nblk(E, T), T, 0, stream>>>(srcI, dstI, deg, nrm, E);

  // ---- layer 1 (Fin = 1): rank-1 updates ----
  k_zero<<<nblk(NF, T), T, 0, stream>>>(acc, NF);
  k_rank1<<<nblk(NF, T), T, 0, stream>>>(acc, x, W1 + 0 * FEAT, NF);  // T0 = x
  k_zero<<<nblk(N, T), T, 0, stream>>>(bufB, N);
  k_prop1<<<nblk(E, T), T, 0, stream>>>(x, srcI, dstI, nrm, bufB, 1.0f, E);
  k_rank1<<<nblk(NF, T), T, 0, stream>>>(acc, bufB, W1 + 1 * FEAT, NF);
  {
    const float* t0 = x;
    const float* t1 = bufB;
    float* wtgt[3] = {bufC, bufA, bufB};
    for (int k = 2; k < 5; ++k) {
      float* t2 = wtgt[k - 2];
      k_negcopy<<<nblk(N, T), T, 0, stream>>>(t2, t0, N);           // t2 = -T0
      k_prop1<<<nblk(E, T), T, 0, stream>>>(t1, srcI, dstI, nrm, t2, 2.0f, E);
      k_rank1<<<nblk(NF, T), T, 0, stream>>>(acc, t2, W1 + k * FEAT, NF);
      t0 = t1; t1 = t2;
    }
  }
  k_bias_relu<<<nblk(NF, T), T, 0, stream>>>(acc, b1, h, NF);

  // ---- layers 2 & 3 (Fin = 64): WMMA GEMMs + scatter props ----
  const float* Wl[2] = {W2, W3};
  const float* bl[2] = {b2, b3};
  for (int L = 0; L < 2; ++L) {
    const float* W = Wl[L];
    k_zero<<<nblk(NF, T), T, 0, stream>>>(acc, NF);
    k_gemm_wmma<<<gemmGrid, 256, 0, stream>>>(h, W + 0 * FEAT * FEAT, acc, N);
    k_zero<<<nblk(NF, T), T, 0, stream>>>(bufB, NF);
    k_propF<<<nblk(EQ, T), T, 0, stream>>>(h, srcI, dstI, nrm, bufB, 1.0f, EQ);
    k_gemm_wmma<<<gemmGrid, 256, 0, stream>>>(bufB, W + 1 * FEAT * FEAT, acc, N);

    const float* t0 = h;
    const float* t1 = bufB;
    float* wtgt[3] = {bufC, bufA, bufB};
    for (int k = 2; k < 5; ++k) {
      float* t2 = wtgt[k - 2];
      k_negcopy<<<nblk(NF, T), T, 0, stream>>>(t2, t0, NF);          // t2 = -T0
      k_propF<<<nblk(EQ, T), T, 0, stream>>>(t1, srcI, dstI, nrm, t2, 2.0f, EQ);
      k_gemm_wmma<<<gemmGrid, 256, 0, stream>>>(t2, W + k * FEAT * FEAT, acc, N);
      t0 = t1; t1 = t2;
    }
    k_bias_relu<<<nblk(NF, T), T, 0, stream>>>(acc, bl[L], h, NF);
  }

  // ---- final FC 64 -> 1 ----
  k_fc<<<nblk(N, T), T, 0, stream>>>(h, Wfc, bfc, (float*)d_out, N);
}
